// MGU_13176959664606
// MI455X (gfx1250) — matrix operand order repaired and checked by reading, not compiled, not run on hardware
//
#include <hip/hip_runtime.h>
#include <hip/hip_bf16.h>

// ---------------------------------------------------------------------------
// MGU recurrent net for MI455X (gfx1250, wave32, WMMA + TDM).
//   Phase 1: xf = x @ Wx_f^T + b_f ; xh = x @ Wx_h^T + b_h  (bf16 WMMA, pipelined)
//   Phase 2: 2048-step scan, H=512 sharded over 16 workgroups; weight slices +
//            per-step activations staged into LDS with tensor_load_to_lds (TDM,
//            TENSORcnt), exchanged via L2, synced with atomic grid barrier +
//            s_cluster_barrier.
// ---------------------------------------------------------------------------

typedef __attribute__((ext_vector_type(16))) __bf16 v16bf;
typedef __attribute__((ext_vector_type(8)))  __bf16 v8bf;
typedef __attribute__((ext_vector_type(8)))  float  v8f;
typedef unsigned int u32;
typedef __attribute__((ext_vector_type(4))) u32 v4u;
typedef __attribute__((ext_vector_type(8))) int  v8i_;
typedef __attribute__((ext_vector_type(4))) int  v4i_;

#define T_STEPS 2048
#define BATCH   64
#define IDIM    512
#define HDIM    512
#define MROWS   (T_STEPS * BATCH)   // 131072
#define NBLK    16                  // scan workgroups (== max cluster size)

#if defined(__has_builtin)
#  if __has_builtin(__builtin_amdgcn_tensor_load_to_lds) && __has_builtin(__builtin_amdgcn_s_wait_tensorcnt)
#    define MGU_HAVE_TDM 1
#  endif
#endif

// ---- WMMA fragment loaders (layouts per CDNA5 ISA 7.12.2) -------
// A 16x32 bf16 (MxK): lanes 0-15 -> K[0..7],[16..23]; lanes 16-31 -> K[8..15],[24..31]
__device__ __forceinline__ v16bf load_a_frag(const __bf16* base, int ld, int lane) {
    const int r    = lane & 15;
    const int koff = (lane >> 4) * 8;
    const __bf16* q = base + r * ld + koff;
    v8bf lo = *(const v8bf*)(q);
    v8bf hi = *(const v8bf*)(q + 16);
    v16bf out;
#pragma unroll
    for (int i = 0; i < 8; ++i) { out[i] = lo[i]; out[i + 8] = hi[i]; }
    return out;
}

// B 32x16 bf16 (KxN), B[k,n] = W[n,k], W row-major [N][K]:
// lanes 0-15 -> K[0..7],[16..23]; lanes 16-31 -> K[8..15],[24..31]
__device__ __forceinline__ v16bf load_b_frag(const __bf16* base, int ld, int lane) {
    const int r    = lane & 15;
    const int koff = (lane >> 4) * 8;
    const __bf16* q = base + r * ld + koff;
    v8bf lo = *(const v8bf*)(q);
    v8bf hi = *(const v8bf*)(q + 16);
    v16bf out;
#pragma unroll
    for (int i = 0; i < 8; ++i) { out[i] = lo[i]; out[i + 8] = hi[i]; }
    return out;
}

__device__ __forceinline__ v8f wmma_bf16(v16bf a, v16bf b, v8f c) {
    return __builtin_amdgcn_wmma_f32_16x16x32_bf16(
        /*neg_a=*/false, a, /*neg_b=*/false, b,
        /*c_mod=*/(short)0, c, /*reuse_a=*/false, /*reuse_b=*/false);
}

#if MGU_HAVE_TDM
// ---- Tensor Data Mover: 2D tile (rows x width_dw dwords) -> LDS ------------
// D# per CDNA5 ISA sec.8. data_size=4B; tile == tensor
// region starting at gaddr with row stride stride_dw dwords. 1D copies use
// rows=1. Wave-level op (EXEC ignored); completion via TENSORcnt.
__device__ __forceinline__ void tdm_load_to_lds(u32 lds_byte_off, const void* gaddr,
                                                u32 width_dw, u32 rows, u32 stride_dw) {
    unsigned long long ga = (unsigned long long)(uintptr_t)gaddr;
    v4u g0;
    g0[0] = 1u;                                   // count=1 (valid), user mode
    g0[1] = lds_byte_off;                         // lds_addr
    g0[2] = (u32)ga;                              // global_addr[31:0]
    g0[3] = (u32)(ga >> 32) | (2u << 30);         // global_addr[56:32] | type=2
    v8i_ g1;
    g1[0] = (int)(2u << 16);                      // wg_mask=0, data_size=2 (4B)
    g1[1] = (int)((width_dw & 0xFFFFu) << 16);                      // tensor_dim0 lo16
    g1[2] = (int)((width_dw >> 16) | ((rows & 0xFFFFu) << 16));     // dim0 hi | dim1 lo
    g1[3] = (int)((rows >> 16) | ((width_dw & 0xFFFFu) << 16));     // dim1 hi | tile_dim0
    g1[4] = (int)(rows & 0xFFFFu);                // tile_dim1 | tile_dim2=0
    g1[5] = (int)stride_dw;                       // tensor_dim0_stride lo32
    g1[6] = 0;                                    // stride hi | dim1_stride lo
    g1[7] = 0;
    v4i_ z4 = {0, 0, 0, 0};
#if __clang_major__ >= 23
    v8i_ z8 = {0, 0, 0, 0, 0, 0, 0, 0};
    __builtin_amdgcn_tensor_load_to_lds(g0, g1, z4, z4, z8, 0);
#else
    __builtin_amdgcn_tensor_load_to_lds(g0, g1, z4, z4, 0);
#endif
}
#endif

// ---- f32 -> bf16 conversion (grid-stride) ----------------------------------
__global__ void mgu_cvt_bf16(const float* __restrict__ in, __bf16* __restrict__ out, long n) {
    long i = (long)blockIdx.x * blockDim.x + threadIdx.x;
    long s = (long)gridDim.x * blockDim.x;
    for (; i < n; i += s) out[i] = (__bf16)in[i];
}

// ---- h0 init: f32 copy into d_out h_final slot + bf16 mirror ---------------
__global__ void mgu_init_h(const float* __restrict__ h0, float* __restrict__ hf,
                           __bf16* __restrict__ hbf) {
    int i = blockIdx.x * blockDim.x + threadIdx.x;
    if (i < BATCH * HDIM) { float v = h0[i]; hf[i] = v; hbf[i] = (__bf16)v; }
}

// ---- Phase 1: input projections  out[m,n] = sum_k x[m,k]*W[n,k] + bias[n] --
// grid (4, MROWS/128, 2), block 256 (8 waves). Wave w -> rows m0+16w, all 128 N.
// Software-pipelined: next A/B fragments load while current WMMA executes.
__global__ void __launch_bounds__(256)
mgu_xproj(const __bf16* __restrict__ xb,
          const __bf16* __restrict__ wf, const __bf16* __restrict__ wh,
          const float* __restrict__ bf, const float* __restrict__ bh,
          float* __restrict__ outf, float* __restrict__ outh) {
    const int sel = blockIdx.z;
    const __bf16* W   = sel ? wh : wf;
    const float* bias = sel ? bh : bf;
    float* out        = sel ? outh : outf;

    const int lane = threadIdx.x & 31;
    const int wave = threadIdx.x >> 5;
    const long m0  = (long)blockIdx.y * 128 + wave * 16;
    const int  n0  = blockIdx.x * 128;

    v8f acc[8];
#pragma unroll
    for (int i = 0; i < 8; ++i) acc[i] = (v8f){0.f,0.f,0.f,0.f,0.f,0.f,0.f,0.f};

    v16bf a = load_a_frag(xb + m0 * IDIM, IDIM, lane);
    for (int k0 = 0; k0 < IDIM; k0 += 32) {
        v16bf a_nxt = a;
        if (k0 + 32 < IDIM) a_nxt = load_a_frag(xb + m0 * IDIM + k0 + 32, IDIM, lane);
        v16bf b = load_b_frag(W + (long)n0 * IDIM + k0, IDIM, lane);
#pragma unroll
        for (int nt = 0; nt < 8; ++nt) {
            v16bf b_nxt = b;
            if (nt < 7)
                b_nxt = load_b_frag(W + (long)(n0 + (nt + 1) * 16) * IDIM + k0, IDIM, lane);
            acc[nt] = wmma_bf16(a, b, acc[nt]);
            b = b_nxt;
        }
        a = a_nxt;
    }

    // C/D layout: n = lane[3:0], m = reg + 8*lane[4]
    const int cn = lane & 15;
    const int cm = (lane >> 4) * 8;
#pragma unroll
    for (int nt = 0; nt < 8; ++nt) {
        const int n = n0 + nt * 16 + cn;
        const float bv = bias[n];
#pragma unroll
        for (int i = 0; i < 8; ++i)
            out[(m0 + cm + i) * HDIM + n] = acc[nt][i] + bv;
    }
}

// ---- grid barrier: atomic counter + CDNA5 cluster barrier ------------------
__device__ __forceinline__ void mgu_grid_barrier(unsigned* ctr, unsigned target) {
    __threadfence();          // release prior global writes
    __syncthreads();
    if (threadIdx.x == 0) {
        atomicAdd(ctr, 1u);
        while (__hip_atomic_load(ctr, __ATOMIC_RELAXED, __HIP_MEMORY_SCOPE_AGENT) < target)
            __builtin_amdgcn_s_sleep(2);
    }
    __syncthreads();
    __threadfence();          // acquire others' writes
    // CDNA5 cluster path: real HW sync when dispatched as a 16-WG cluster,
    // architecturally NOP otherwise (ClusterID==0).
    __builtin_amdgcn_s_cluster_barrier();
}

// ---- Phase 2: recurrent scan. 16 blocks x 256 threads, dyn LDS 192KB -------
// Block b owns N-slice [32b, 32b+32). Wave tile: 4 M-tiles x 2 N-tiles.
// LDS byte layout (no static LDS -> dynamic base offset 0):
//   [0,32K)   Wf slice   [32K,64K) Wh slice   [64K,128K) hA   [128K,192K) uA
#define LDS_WF_OFF 0u
#define LDS_WH_OFF 32768u
#define LDS_HA_OFF 65536u
#define LDS_UA_OFF 131072u

__global__ void __launch_bounds__(256)
mgu_scan(const float* __restrict__ xf,   // [T*B*H] (aliases y region; read-then-overwrite)
         const float* __restrict__ xh,   // [T*B*H]
         const __bf16* __restrict__ whf, // [H][H] bf16 row-major (n,k)
         const __bf16* __restrict__ whh,
         float* __restrict__ y,          // d_out [T*B*H]
         float* __restrict__ hf,         // d_out + T*B*H, running h (f32)
         __bf16* __restrict__ hbf,       // [B*H]
         __bf16* __restrict__ ubf,       // [2][B*H] double-buffered
         unsigned* __restrict__ ctr) {
    extern __shared__ __bf16 lds[];
    __bf16* Wf = lds;                    // [32][512]  32KB
    __bf16* Wh = lds + 32 * HDIM;        // [32][512]  32KB
    __bf16* hA = lds + 64 * HDIM;        // [64][512]  64KB
    __bf16* uA = lds + 128 * HDIM;       // [64][512]  64KB

    const int tid  = threadIdx.x;
    const int lane = tid & 31;
    const int wave = tid >> 5;
    const int n0   = blockIdx.x * 32;    // global N offset of this block

    // Stage resident weight slices (rows n0..n0+31, full K) via TDM
#if MGU_HAVE_TDM
    if (wave == 0) {
        tdm_load_to_lds(LDS_WF_OFF, whf + (long)n0 * HDIM, HDIM / 2, 32, HDIM / 2);
        tdm_load_to_lds(LDS_WH_OFF, whh + (long)n0 * HDIM, HDIM / 2, 32, HDIM / 2);
        __builtin_amdgcn_s_wait_tensorcnt(0);
    }
#else
    for (int idx = tid; idx < 32 * HDIM; idx += 256) {
        const int r = idx >> 9, k = idx & (HDIM - 1);
        Wf[idx] = whf[(long)(n0 + r) * HDIM + k];
        Wh[idx] = whh[(long)(n0 + r) * HDIM + k];
    }
#endif

    const int mt = wave & 3;             // batch tile 0..3
    const int nt = wave >> 2;            // local col tile 0..1
    const int m_base = mt * 16;
    const int nl0 = nt * 16;
    const int n_g = n0 + nl0;
    const int cn = lane & 15;
    const int cm = (lane >> 4) * 8;

    unsigned ep = 0;
    __syncthreads();

    for (int t = 0; t < T_STEPS; ++t) {
        __bf16* u_cur = ubf + (t & 1) * (BATCH * HDIM);
        const long tbase = (long)t * BATCH * HDIM;
        const int hidx0 = (m_base + cm) * HDIM + n_g + cn;

        // 1. stage full h (bf16) into LDS (64KB contiguous -> TDM 1D tile)
#if MGU_HAVE_TDM
        if (wave == 0) {
            tdm_load_to_lds(LDS_HA_OFF, hbf, BATCH * HDIM / 2, 1, BATCH * HDIM / 2);
            __builtin_amdgcn_s_wait_tensorcnt(0);
        }
#else
        for (int idx = tid; idx < BATCH * HDIM; idx += 256) hA[idx] = hbf[idx];
#endif
        // prefetch next step's gate inputs while TDM runs
        if (t + 1 < T_STEPS) {
            __builtin_prefetch(xf + tbase + BATCH * HDIM + hidx0, 0, 3);
            __builtin_prefetch(xh + tbase + BATCH * HDIM + hidx0, 0, 3);
        }
        __syncthreads();

        // 2. pre_f tile = h @ Wh_f^T  (K=512 -> 16 WMMA)
        v8f cfr = (v8f){0.f,0.f,0.f,0.f,0.f,0.f,0.f,0.f};
#pragma unroll
        for (int k0 = 0; k0 < HDIM; k0 += 32) {
            v16bf a = load_a_frag(hA + m_base * HDIM + k0, HDIM, lane);
            v16bf b = load_b_frag(Wf + nl0 * HDIM + k0, HDIM, lane);
            cfr = wmma_bf16(a, b, cfr);
        }

        // 3. f = sigmoid(pre_f + xf[t]);  u = f*h  -> global (bf16, dbl-buffered)
        float fv[8], hv[8];
#pragma unroll
        for (int i = 0; i < 8; ++i) {
            const int hidx = hidx0 + i * HDIM;
            const float pf = cfr[i] + xf[tbase + hidx];
            const float f  = 1.f / (1.f + __expf(-pf));
            const float h  = hf[hidx];
            fv[i] = f; hv[i] = h;
            u_cur[hidx] = (__bf16)(f * h);
        }
        mgu_grid_barrier(ctr, (++ep) * NBLK);

        // 4. stage full u via TDM, pre_h tile = u @ Wh_h^T
#if MGU_HAVE_TDM
        if (wave == 0) {
            tdm_load_to_lds(LDS_UA_OFF, u_cur, BATCH * HDIM / 2, 1, BATCH * HDIM / 2);
            __builtin_amdgcn_s_wait_tensorcnt(0);
        }
#else
        for (int idx = tid; idx < BATCH * HDIM; idx += 256) uA[idx] = u_cur[idx];
#endif
        __syncthreads();
        v8f chr = (v8f){0.f,0.f,0.f,0.f,0.f,0.f,0.f,0.f};
#pragma unroll
        for (int k0 = 0; k0 < HDIM; k0 += 32) {
            v16bf a = load_a_frag(uA + m_base * HDIM + k0, HDIM, lane);
            v16bf b = load_b_frag(Wh + nl0 * HDIM + k0, HDIM, lane);
            chr = wmma_bf16(a, b, chr);
        }

        // 5. h_new = (1-f)*h + f*tanh(pre_h + xh[t]); emit y, h, h_bf16
#pragma unroll
        for (int i = 0; i < 8; ++i) {
            const int hidx = hidx0 + i * HDIM;
            const float ht = tanhf(chr[i] + xh[tbase + hidx]);
            const float hn = (1.f - fv[i]) * hv[i] + fv[i] * ht;
            y[tbase + hidx] = hn;
            hf[hidx]  = hn;
            hbf[hidx] = (__bf16)hn;
        }
        mgu_grid_barrier(ctr, (++ep) * NBLK);
    }
}

// ---------------------------------------------------------------------------
extern "C" void kernel_launch(void* const* d_in, const int* in_sizes, int n_in,
                              void* d_out, int out_size, void* d_ws, size_t ws_size,
                              hipStream_t stream) {
    // inputs: x_seq, h0, Wx_f, Wh_f, b_f, Wx_h, Wh_h, b_h
    const float* x_seq = (const float*)d_in[0];
    const float* h0    = (const float*)d_in[1];
    const float* Wx_f  = (const float*)d_in[2];
    const float* Wh_f  = (const float*)d_in[3];
    const float* b_f   = (const float*)d_in[4];
    const float* Wx_h  = (const float*)d_in[5];
    const float* Wh_h  = (const float*)d_in[6];
    const float* b_h   = (const float*)d_in[7];

    float* y  = (float*)d_out;                      // [T,B,H]
    float* hf = y + (long)T_STEPS * BATCH * HDIM;   // [B,H] h_final slot

    // workspace layout
    char* w = (char*)d_ws;
    __bf16* x_bf  = (__bf16*)w;                                   // 128 MB
    float*  xh    = (float*)(w + 134217728L);                     // 256 MB
    __bf16* wxf_b = (__bf16*)(w + 402653184L);                    // 4x 512KB
    __bf16* wxh_b = wxf_b + (long)HDIM * IDIM;
    __bf16* whf_b = wxh_b + (long)HDIM * IDIM;
    __bf16* whh_b = whf_b + (long)HDIM * HDIM;
    __bf16* h_bf  = whh_b + (long)HDIM * HDIM;                    // 64 KB
    __bf16* u_bf  = h_bf + BATCH * HDIM;                          // 2x 64 KB
    unsigned* ctr = (unsigned*)(w + 402653184L + 2097152L + 65536L + 131072L);

    // xf is materialized into the y region of d_out (read at step t before y[t] write)
    float* xf = y;

    hipMemsetAsync(ctr, 0, sizeof(unsigned), stream);

    // bf16 conversions
    mgu_cvt_bf16<<<4096, 256, 0, stream>>>(x_seq, x_bf, (long)MROWS * IDIM);
    mgu_cvt_bf16<<<256, 256, 0, stream>>>(Wx_f, wxf_b, (long)HDIM * IDIM);
    mgu_cvt_bf16<<<256, 256, 0, stream>>>(Wx_h, wxh_b, (long)HDIM * IDIM);
    mgu_cvt_bf16<<<256, 256, 0, stream>>>(Wh_f, whf_b, (long)HDIM * HDIM);
    mgu_cvt_bf16<<<256, 256, 0, stream>>>(Wh_h, whh_b, (long)HDIM * HDIM);

    mgu_init_h<<<(BATCH * HDIM + 255) / 256, 256, 0, stream>>>(h0, hf, h_bf);

    // Phase 1: input projections (bandwidth-bound; bf16 WMMA)
    dim3 pgrid(HDIM / 128, MROWS / 128, 2);
    mgu_xproj<<<pgrid, 256, 0, stream>>>(x_bf, wxf_b, wxh_b, b_f, b_h, xf, xh);

    // Phase 2: recurrent scan, 16 persistent workgroups, 192 KB dynamic LDS each
    mgu_scan<<<NBLK, 256, 196608, stream>>>(xf, xh, whf_b, whh_b,
                                            y, hf, h_bf, u_bf, ctr);
}
